// TransferTraj_38749194944628
// MI455X (gfx1250) — matrix operand order repaired and checked by reading, not compiled, not run on hardware
//
#include <hip/hip_runtime.h>

typedef _Float16 v16h __attribute__((ext_vector_type(16)));
typedef _Float16 v8h  __attribute__((ext_vector_type(8)));
typedef float    v8f  __attribute__((ext_vector_type(8)));

#define BB 4
#define LL 512
#define NQ (BB*LL)          // 2048
#define EE 128
#define DD 256
#define NPOI 20000
#define NROAD 10000
#define KPAD_POI 20032      // padded to multiple of 64
#define KPAD_ROAD 10048     // padded to multiple of 64

static __device__ __forceinline__ v8f wmma_f16(v16h a, v16h b, v8f c) {
    return __builtin_amdgcn_wmma_f32_16x16x32_f16(false, a, false, b, (short)0, c,
                                                  false, false);
}

// ---- transpose+convert weight (E x D f32) -> (D x E f16) --------------------
__global__ __launch_bounds__(256) void k_prep_w(const float* __restrict__ w,
                                                _Float16* __restrict__ wT) {
    int idx = blockIdx.x * 256 + threadIdx.x;     // 0 .. 32767
    int d = idx >> 7, e = idx & 127;
    wT[d * EE + e] = (_Float16)w[e * DD + d];
}

// ---- zero the K-padding region of a transposed table ------------------------
// grid covers DD * (kpad - ntab) elements
__global__ __launch_bounds__(256) void k_zero_pad(_Float16* __restrict__ tabT,
                                                  int ntab, int kpad) {
    int padw = kpad - ntab;
    int idx = blockIdx.x * 256 + threadIdx.x;
    int d = idx / padw, j = idx - d * padw;
    if (d < DD) tabT[(size_t)d * kpad + ntab + j] = (_Float16)0.f;
}

// ---- token table: LN(tok_emb) @ tok_w + tok_b  (6 x 256) --------------------
__global__ __launch_bounds__(256) void k_tok_table(const float* __restrict__ tok_emb,
                                                   const float* __restrict__ g,
                                                   const float* __restrict__ b,
                                                   const float* __restrict__ w,
                                                   const float* __restrict__ bias,
                                                   float* __restrict__ table) {
    __shared__ float sT[6][EE];
    int t = threadIdx.x;
    if (t < 6) {
        float mu = 0.f, s2 = 0.f;
        for (int e = 0; e < EE; ++e) mu += tok_emb[t * EE + e];
        mu *= (1.f / EE);
        for (int e = 0; e < EE; ++e) {
            float x = tok_emb[t * EE + e] - mu;
            s2 += x * x;
        }
        s2 *= (1.f / EE);
        float rs = rsqrtf(s2 + 1e-5f);
        for (int e = 0; e < EE; ++e)
            sT[t][e] = (tok_emb[t * EE + e] - mu) * rs * g[e] + b[e];
    }
    __syncthreads();
    int d = t;
    for (int r = 0; r < 6; ++r) {
        float acc = bias[d];
        for (int e = 0; e < EE; ++e) acc += sT[r][e] * w[e * DD + d];
        table[r * DD + d] = acc;
    }
}

// ---- embed producer: rows of LN(src) @ w + bias -> transposed f16 table -----
// block = 256 threads (8 waves), 16 table rows x all 256 output cols.
__global__ __launch_bounds__(256) void k_embed(const float* __restrict__ src,
                                               const float* __restrict__ g,
                                               const float* __restrict__ bvec,
                                               const _Float16* __restrict__ wT,
                                               const float* __restrict__ bias,
                                               _Float16* __restrict__ dstT,
                                               int kpad) {
    __shared__ alignas(32) _Float16 sA[16][EE];
    __shared__ alignas(32) _Float16 sOut[16][DD];
    int t = threadIdx.x;
    int r0 = blockIdx.x * 16;

    // LayerNorm: thread t handles row (t>>4), elements (t&15)*8 .. +7
    int rl = t >> 4, seg = t & 15;
    const float* row = src + (size_t)(r0 + rl) * EE + seg * 8;
    float x[8], s = 0.f, s2 = 0.f;
#pragma unroll
    for (int j = 0; j < 8; ++j) { x[j] = row[j]; s += x[j]; s2 += x[j] * x[j]; }
    for (int m = 1; m < 16; m <<= 1) {
        s  += __shfl_xor(s, m, 16);
        s2 += __shfl_xor(s2, m, 16);
    }
    float mu  = s * (1.f / EE);
    float var = s2 * (1.f / EE) - mu * mu;
    float rs  = rsqrtf(var + 1e-5f);
#pragma unroll
    for (int j = 0; j < 8; ++j) {
        int e = seg * 8 + j;
        sA[rl][e] = (_Float16)((x[j] - mu) * rs * g[e] + bvec[e]);
    }
    __syncthreads();

    int w = t >> 5, L = t & 31, sel = L >> 4, lc = L & 15;
    int col0 = w * 32 + lc, col1 = col0 + 16;
    v8f acc0 = {0.f, 0.f, 0.f, 0.f, 0.f, 0.f, 0.f, 0.f};
    v8f acc1 = acc0;
#pragma unroll
    for (int k0 = 0; k0 < EE; k0 += 32) {
        union { v16h v; v8h h[2]; } a;
        a.h[0] = *(const v8h*)&sA[lc][k0 + sel * 16];
        a.h[1] = *(const v8h*)&sA[lc][k0 + sel * 16 + 8];
        v16h b0 = *(const v16h*)&wT[(size_t)col0 * EE + k0 + sel * 16];
        v16h b1 = *(const v16h*)&wT[(size_t)col1 * EE + k0 + sel * 16];
        acc0 = wmma_f16(a.v, b0, acc0);
        acc1 = wmma_f16(a.v, b1, acc1);
    }
    float bi0 = bias[col0], bi1 = bias[col1];
#pragma unroll
    for (int i = 0; i < 8; ++i) {
        int m = sel * 8 + i;
        sOut[m][col0] = (_Float16)(acc0[i] + bi0);
        sOut[m][col1] = (_Float16)(acc1[i] + bi1);
    }
    __syncthreads();

    // transposed store: thread t == output column d; 16 rows contiguous in dstT
    union { v16h v; _Float16 e[16]; } o;
#pragma unroll
    for (int i = 0; i < 16; ++i) o.e[i] = sOut[i][t];
    *(v16h*)&dstT[(size_t)t * kpad + r0] = o.v;
}

// ---- radius-mean as masked WMMA GEMM ----------------------------------------
// block = 256 threads (8 waves), 32 query rows x full D=256.
// K-step = 64, mask tile double-buffered in LDS (one barrier per step).
__global__ __launch_bounds__(256) void k_radius(const float* __restrict__ spatial,
                                                const float* __restrict__ fp,
                                                const int* __restrict__ token,
                                                const float* __restrict__ coors,
                                                const _Float16* __restrict__ tabT,
                                                const float* __restrict__ tokTab,
                                                float* __restrict__ out,
                                                int ntab, int kpad, float thresh,
                                                int ch, int apply_fmask) {
    __shared__ alignas(32) _Float16 sA[2][32][64];
    __shared__ float sCnt[256];
    __shared__ float sInv[32];
    __shared__ int   sTok[32];
    __shared__ int   sZero[32];

    int t  = threadIdx.x;
    int q0 = blockIdx.x * 32;
    int r  = t & 31;                  // mask row 0..31
    int q  = q0 + r;
    int bi = q >> 9;                  // q / 512
    float qx = spatial[q * 2 + 0] + fp[bi * 2 + 0];
    float qy = spatial[q * 2 + 1] + fp[bi * 2 + 1];
    float qq = qx * qx + qy * qy;
    int kloc = (t >> 5) * 8;          // this thread's 8-wide K slice

    int w = t >> 5, L = t & 31, sel = L >> 4, lc = L & 15;
    int col0 = w * 32 + lc, col1 = col0 + 16;
    v8f acc00 = {0.f, 0.f, 0.f, 0.f, 0.f, 0.f, 0.f, 0.f};
    v8f acc01 = acc00, acc10 = acc00, acc11 = acc00;
    float cnt = 0.f;

    int nsteps = kpad >> 6;
    for (int s = 0; s < nsteps; ++s) {
        int k0 = s << 6;
        int p  = s & 1;

        // ---- compute 8 masks for (row r, k = k0+kloc .. +7) -----------------
        union { v8h v; _Float16 e[8]; } mv;
#pragma unroll
        for (int jj = 0; jj < 2; ++jj) {
            int k = k0 + kloc + jj * 4;
            float px[4], py[4];
            if (k + 3 < ntab) {
                float4 v0 = *(const float4*)&coors[2 * k];
                float4 v1 = *(const float4*)&coors[2 * k + 4];
                px[0] = v0.x; py[0] = v0.y; px[1] = v0.z; py[1] = v0.w;
                px[2] = v1.x; py[2] = v1.y; px[3] = v1.z; py[3] = v1.w;
            } else {
#pragma unroll
                for (int u = 0; u < 4; ++u) {
                    px[u] = (k + u < ntab) ? coors[2 * (k + u)]     : 3e8f;
                    py[u] = (k + u < ntab) ? coors[2 * (k + u) + 1] : 3e8f;
                }
            }
#pragma unroll
            for (int u = 0; u < 4; ++u) {
                // same expanded-form d^2 as the reference einsum
                float d2 = qq + (px[u] * px[u] + py[u] * py[u])
                              - 2.f * (qx * px[u] + qy * py[u]);
                float m = (d2 < thresh) ? 1.f : 0.f;
                cnt += m;
                mv.e[jj * 4 + u] = (_Float16)m;
            }
        }
        *(v8h*)&sA[p][r][kloc] = mv.v;      // one 16B ds store
        __syncthreads();                     // single barrier (double-buffered)

        // ---- consume: 2 K-substeps x 2 M-tiles x 2 N-tiles = 8 WMMAs --------
#pragma unroll
        for (int ks = 0; ks < 2; ++ks) {
            int kk = ks * 32;
            union { v16h v; v8h h[2]; } a0, a1;
            a0.h[0] = *(const v8h*)&sA[p][lc][kk + sel * 16];
            a0.h[1] = *(const v8h*)&sA[p][lc][kk + sel * 16 + 8];
            a1.h[0] = *(const v8h*)&sA[p][16 + lc][kk + sel * 16];
            a1.h[1] = *(const v8h*)&sA[p][16 + lc][kk + sel * 16 + 8];
            const _Float16* bp0 = &tabT[(size_t)col0 * kpad + k0 + kk + sel * 16];
            const _Float16* bp1 = &tabT[(size_t)col1 * kpad + k0 + kk + sel * 16];
            __builtin_prefetch(bp0 + 64, 0, 1);   // next K-step, same lane row
            v16h b0 = *(const v16h*)bp0;
            v16h b1 = *(const v16h*)bp1;
            acc00 = wmma_f16(a0.v, b0, acc00);
            acc01 = wmma_f16(a0.v, b1, acc01);
            acc10 = wmma_f16(a1.v, b0, acc10);
            acc11 = wmma_f16(a1.v, b1, acc11);
        }
    }

    // ---- per-row counts ------------------------------------------------------
    sCnt[t] = cnt;
    __syncthreads();
    if (t < 32) {
        float c = 0.f;
        for (int i = 0; i < 8; ++i) c += sCnt[t + 32 * i];
        c = fmaxf(c, 1.f);
        sInv[t] = 1.f / c;
        int tok0 = token[(q0 + t) * 2];
        sTok[t]  = tok0;
        sZero[t] = !((tok0 == 0) || (tok0 == 1));
    }
    __syncthreads();

    // ---- epilogue: divide, fmask, +tok_e0, store -----------------------------
#pragma unroll
    for (int i = 0; i < 8; ++i) {
        int m0r = sel * 8 + i;
        int m1r = 16 + m0r;
        float i0 = sInv[m0r], i1 = sInv[m1r];
        float v00 = acc00[i] * i0, v01 = acc01[i] * i0;
        float v10 = acc10[i] * i1, v11 = acc11[i] * i1;
        if (apply_fmask) {
            if (sZero[m0r]) { v00 = 0.f; v01 = 0.f; }
            if (sZero[m1r]) { v10 = 0.f; v11 = 0.f; }
        }
        v00 += tokTab[sTok[m0r] * DD + col0];
        v01 += tokTab[sTok[m0r] * DD + col1];
        v10 += tokTab[sTok[m1r] * DD + col0];
        v11 += tokTab[sTok[m1r] * DD + col1];
        out[((size_t)(q0 + m0r) * 3 + ch) * DD + col0] = v00;
        out[((size_t)(q0 + m0r) * 3 + ch) * DD + col1] = v01;
        out[((size_t)(q0 + m1r) * 3 + ch) * DD + col0] = v10;
        out[((size_t)(q0 + m1r) * 3 + ch) * DD + col1] = v11;
    }
}

// ---- spatial branch: leaky_relu(spatial@w1+b1)@w2 + b2 + tok_e0 -------------
__global__ __launch_bounds__(256) void k_spatial(const float* __restrict__ spatial,
                                                 const int* __restrict__ token,
                                                 const float* __restrict__ w1,
                                                 const float* __restrict__ b1,
                                                 const float* __restrict__ w2,
                                                 const float* __restrict__ b2,
                                                 const float* __restrict__ tokTab,
                                                 float* __restrict__ out) {
    __shared__ float sH[EE];
    int q = blockIdx.x;
    int t = threadIdx.x;
    float qx = spatial[q * 2], qy = spatial[q * 2 + 1];
    if (t < EE) {
        float h = qx * w1[t] + qy * w1[EE + t] + b1[t];
        sH[t] = h > 0.f ? h : 0.01f * h;
    }
    __syncthreads();
    int tok0 = token[q * 2];
    float acc = b2[t];
    for (int e = 0; e < EE; ++e) acc += sH[e] * w2[e * DD + t];
    out[((size_t)q * 3 + 0) * DD + t] = acc + tokTab[tok0 * DD + t];
}

extern "C" void kernel_launch(void* const* d_in, const int* in_sizes, int n_in,
                              void* d_out, int out_size, void* d_ws, size_t ws_size,
                              hipStream_t stream) {
    const float* spatial     = (const float*)d_in[0];
    const float* first_point = (const float*)d_in[1];
    const int*   token       = (const int*)d_in[2];
    const float* poi_coors   = (const float*)d_in[3];
    const float* poi_embed   = (const float*)d_in[4];
    const float* road_coors  = (const float*)d_in[5];
    const float* road_embed  = (const float*)d_in[6];
    const float* w_sp1       = (const float*)d_in[7];
    const float* b_sp1       = (const float*)d_in[8];
    const float* w_sp2       = (const float*)d_in[9];
    const float* b_sp2       = (const float*)d_in[10];
    const float* poi_ln_g    = (const float*)d_in[11];
    const float* poi_ln_b    = (const float*)d_in[12];
    const float* poi_w       = (const float*)d_in[13];
    const float* poi_b       = (const float*)d_in[14];
    const float* road_ln_g   = (const float*)d_in[15];
    const float* road_ln_b   = (const float*)d_in[16];
    const float* road_w      = (const float*)d_in[17];
    const float* road_b      = (const float*)d_in[18];
    const float* tok_emb     = (const float*)d_in[19];
    const float* tok_ln_g    = (const float*)d_in[20];
    const float* tok_ln_b    = (const float*)d_in[21];
    const float* tok_w       = (const float*)d_in[22];
    const float* tok_b       = (const float*)d_in[23];
    float* out = (float*)d_out;

    char* ws = (char*)d_ws;
    float*    tokTab = (float*)(ws + 0);                         //  6144 B
    _Float16* wTpoi  = (_Float16*)(ws + 8192);                   // 65536 B
    _Float16* wTroad = (_Float16*)(ws + 8192 + 65536);           // 65536 B
    _Float16* poiT   = (_Float16*)(ws + 139264);                 // 256*20032*2
    _Float16* roadT  = (_Float16*)(ws + 139264 + 10256384);      // 256*10048*2

    // stage 0: weight transposes + pad zero-fill + token table
    k_prep_w<<<(DD * EE) / 256, 256, 0, stream>>>(poi_w, wTpoi);
    k_prep_w<<<(DD * EE) / 256, 256, 0, stream>>>(road_w, wTroad);
    k_zero_pad<<<(DD * (KPAD_POI - NPOI) + 255) / 256, 256, 0, stream>>>(
        poiT, NPOI, KPAD_POI);
    k_zero_pad<<<(DD * (KPAD_ROAD - NROAD) + 255) / 256, 256, 0, stream>>>(
        roadT, NROAD, KPAD_ROAD);
    k_tok_table<<<1, 256, 0, stream>>>(tok_emb, tok_ln_g, tok_ln_b, tok_w, tok_b,
                                       tokTab);

    // stage 1: LN + GEMM producing transposed f16 embed tables (WMMA)
    k_embed<<<NPOI / 16, 256, 0, stream>>>(poi_embed, poi_ln_g, poi_ln_b, wTpoi,
                                           poi_b, poiT, KPAD_POI);
    k_embed<<<NROAD / 16, 256, 0, stream>>>(road_embed, road_ln_g, road_ln_b,
                                            wTroad, road_b, roadT, KPAD_ROAD);

    // stage 2: masked-GEMM radius means (WMMA), channels 1 and 2
    k_radius<<<NQ / 32, 256, 0, stream>>>(spatial, first_point, token, poi_coors,
                                          poiT, tokTab, out, NPOI, KPAD_POI,
                                          25.0f, 1, 1);
    k_radius<<<NQ / 32, 256, 0, stream>>>(spatial, first_point, token, road_coors,
                                          roadT, tokTab, out, NROAD, KPAD_ROAD,
                                          25.0f, 2, 0);

    // stage 3: spatial branch, channel 0
    k_spatial<<<NQ, 256, 0, stream>>>(spatial, token, w_sp1, b_sp1, w_sp2, b_sp2,
                                      tokTab, out);
}
